// TreeLSTMCell_33638183863176
// MI455X (gfx1250) — matrix operand order repaired and checked by reading, not compiled, not run on hardware
//
#include <hip/hip_runtime.h>

#define HS 256
#define XS 256
#define NN 262144
#define NHOFF ((size_t)NN * (size_t)HS)

// Pre-swizzled bf16 B-fragment storage:
//   leaf:     48 tiles * 8  ksteps = 384  blocks of 1024 B = 393216 B
//   internal: 80 tiles * 16 ksteps = 1280 blocks of 1024 B = 1310720 B
#define WS_LEAF_BYTES  (384u * 1024u)
#define WS_INT_BYTES   (1280u * 1024u)
#define WS_NEEDED      ((size_t)WS_LEAF_BYTES + (size_t)WS_INT_BYTES)

typedef __attribute__((ext_vector_type(16))) __bf16 v16bf;
typedef __attribute__((ext_vector_type(8)))  __bf16 v8bf;
typedef __attribute__((ext_vector_type(2)))  __bf16 bf16x2;
typedef __attribute__((ext_vector_type(8)))  float  v8f;

#define WMMA_BF16(A, B, C) \
  __builtin_amdgcn_wmma_f32_16x16x32_bf16(false, (A), false, (B), (short)0, (C), false, false)

static __device__ __forceinline__ float fast_rcp(float x) { return __builtin_amdgcn_rcpf(x); }
static __device__ __forceinline__ float sigm(float x) { return fast_rcp(1.0f + __expf(-x)); }
static __device__ __forceinline__ float tanh_(float x) {
  float xc = fminf(fmaxf(x, -15.0f), 15.0f);
  float e = __expf(2.0f * xc);
  return (e - 1.0f) * fast_rcp(e + 1.0f);
}

static __device__ __forceinline__ bf16x2 cvt2(float a, float b) {
  bf16x2 r; r.x = (__bf16)a; r.y = (__bf16)b; return r;
}

// Build a B-fragment (contiguous-K order as in load_a_lds: lane = (col, khalf), 16
// consecutive K values per lane) from 16 consecutive f32 weights W[n][k..k+15].
static __device__ __forceinline__ v16bf load_b_f32(const float* __restrict__ p) {
  const float4* q = (const float4*)p;
  float4 a = q[0], b = q[1], c = q[2], d = q[3];
  v16bf r;
  r[0]  = (__bf16)a.x; r[1]  = (__bf16)a.y; r[2]  = (__bf16)a.z; r[3]  = (__bf16)a.w;
  r[4]  = (__bf16)b.x; r[5]  = (__bf16)b.y; r[6]  = (__bf16)b.z; r[7]  = (__bf16)b.w;
  r[8]  = (__bf16)c.x; r[9]  = (__bf16)c.y; r[10] = (__bf16)c.z; r[11] = (__bf16)c.w;
  r[12] = (__bf16)d.x; r[13] = (__bf16)d.y; r[14] = (__bf16)d.z; r[15] = (__bf16)d.w;
  return r;
}

// 16-bit A-fragment (contiguous K): base = &row[kb*16]; per k-step the two
// halves sit at +ks*32 and +ks*32+8 elements -> ds_load immediates.
static __device__ __forceinline__ v16bf load_a_lds(const __bf16* base, int ksoff) {
  v8bf lo = *(const v8bf*)(base + ksoff);
  v8bf hi = *(const v8bf*)(base + ksoff + 8);
  return __builtin_shufflevector(lo, hi, 0, 1, 2, 3, 4, 5, 6, 7, 8, 9, 10, 11, 12, 13, 14, 15);
}

// ---------------------------------------------------------------------------
// Weight pre-swizzle: f32 row-major -> bf16 B-fragments, one 32B chunk per
// (tile, kstep, lane). GEMM loop then does pure coalesced b128 loads.
// ---------------------------------------------------------------------------
__global__ __launch_bounds__(256) void swizzle_leaf(const float* __restrict__ W,
                                                    __bf16* __restrict__ dst) {
  const int t = blockIdx.x * 256 + threadIdx.x;  // 384 blocks * 32 lanes = 12288
  const int lane = t & 31, blk = t >> 5;
  const int tile = blk >> 3, ks = blk & 7;
  const int n = tile * 16 + (lane & 15);
  const int k = ks * 32 + (lane >> 4) * 16;
  v16bf r = load_b_f32(W + (size_t)n * XS + k);
  __bf16* d = dst + (size_t)t * 16;
  *(v8bf*)(d)     = __builtin_shufflevector(r, r, 0, 1, 2, 3, 4, 5, 6, 7);
  *(v8bf*)(d + 8) = __builtin_shufflevector(r, r, 8, 9, 10, 11, 12, 13, 14, 15);
}

__global__ __launch_bounds__(256) void swizzle_internal(const float* __restrict__ Ufw,
                                                        const float* __restrict__ Uiw,
                                                        __bf16* __restrict__ dst) {
  const int t = blockIdx.x * 256 + threadIdx.x;  // 1280 blocks * 32 lanes = 40960
  const int lane = t & 31, blk = t >> 5;
  const int tile = blk >> 4, ks = blk & 15;
  const int n = tile * 16 + (lane & 15);
  const int k = ks * 32 + (lane >> 4) * 16;
  const float* row = (n < 512) ? (Ufw + (size_t)n * (2 * HS))
                               : (Uiw + (size_t)(n - 512) * (2 * HS));
  v16bf r = load_b_f32(row + k);
  __bf16* d = dst + (size_t)t * 16;
  *(v8bf*)(d)     = __builtin_shufflevector(r, r, 0, 1, 2, 3, 4, 5, 6, 7);
  *(v8bf*)(d + 8) = __builtin_shufflevector(r, r, 8, 9, 10, 11, 12, 13, 14, 15);
}

// ---------------------------------------------------------------------------
// Leaf path: iou = x @ W_iou^T + b ; c = sig(i)*tanh(u) ; h = sig(o)*tanh(c)
// Block: 32 rows, 8 waves. Wave owns column groups g = 2w, 2w+1; group g needs
// GEMM tiles at n = g*16 + {0,256,512} (i,o,u) -> register-local epilogue.
// Tile order: t = ch*2 + gi.
// ---------------------------------------------------------------------------
template <bool USE_WS>
__global__ __launch_bounds__(256) void treelstm_leaf(
    const float* __restrict__ x, const float* __restrict__ Ww, const float* __restrict__ Wb,
    const __bf16* __restrict__ wsB,
    float* __restrict__ outH, float* __restrict__ outC) {
  __shared__ __attribute__((aligned(16))) __bf16 sA[32][XS + 8];

  const int tid = threadIdx.x;
  const int m0 = blockIdx.x * 32;

  {
    const int r = tid >> 3, seg = tid & 7;
    const float4* src = (const float4*)(x + (size_t)(m0 + r) * XS + seg * 32);
    bf16x2* dst = (bf16x2*)(&sA[r][seg * 32]);
#pragma unroll
    for (int q = 0; q < 8; ++q) {
      float4 f = src[q];
      dst[2 * q]     = cvt2(f.x, f.y);
      dst[2 * q + 1] = cvt2(f.z, f.w);
    }
  }
  __syncthreads();

  const int wave = tid >> 5, lane = tid & 31;
  const int nrow = lane & 15, kb = lane >> 4;

  // Hoisted per-tile B base pointers (advance by immediate ks*1024B / ks*128B).
  const __bf16* bp16[6] = {};
  const float*  bp32[6] = {};
#pragma unroll
  for (int t = 0; t < 6; ++t) {
    const int gi = t & 1, ch = t >> 1;
    if (USE_WS) {
      const int tile = (wave * 2 + gi) + ch * 16;
      bp16[t] = wsB + ((size_t)tile * 8 * 32 + lane) * 16;
    } else {
      const int n = (wave * 2 + gi) * 16 + ch * 256 + nrow;
      bp32[t] = Ww + (size_t)n * XS + kb * 16;
    }
  }
  const __bf16* aBase0 = &sA[nrow][kb * 16];
  const __bf16* aBase1 = &sA[nrow + 16][kb * 16];

  v8f zero;
#pragma unroll
  for (int i = 0; i < 8; ++i) zero[i] = 0.0f;
  v8f acc[6][2];
#pragma unroll
  for (int t = 0; t < 6; ++t) {
    acc[t][0] = zero;
    acc[t][1] = zero;
  }

#pragma unroll
  for (int ks = 0; ks < XS / 32; ++ks) {
    v16bf a0 = load_a_lds(aBase0, ks * 32);
    v16bf a1 = load_a_lds(aBase1, ks * 32);
#pragma unroll
    for (int t = 0; t < 6; ++t) {
      v16bf b;
      if (USE_WS) b = *(const v16bf*)(bp16[t] + ks * 512);
      else        b = load_b_f32(bp32[t] + ks * 32);
      acc[t][0] = WMMA_BF16(a0, b, acc[t][0]);
      acc[t][1] = WMMA_BF16(a1, b, acc[t][1]);
    }
  }

#pragma unroll
  for (int gi = 0; gi < 2; ++gi) {
    const int col = (wave * 2 + gi) * 16 + nrow;
    const float bI = Wb[col], bO = Wb[HS + col], bU = Wb[2 * HS + col];
#pragma unroll
    for (int mt = 0; mt < 2; ++mt) {
#pragma unroll
      for (int v = 0; v < 8; ++v) {
        const int row = m0 + mt * 16 + v + 8 * kb;
        float cc = sigm(acc[0 * 2 + gi][mt][v] + bI) * tanh_(acc[2 * 2 + gi][mt][v] + bU);
        float hh = sigm(acc[1 * 2 + gi][mt][v] + bO) * tanh_(cc);
        outH[(size_t)row * HS + col] = hh;
        outC[(size_t)row * HS + col] = cc;
      }
    }
  }
}

// ---------------------------------------------------------------------------
// Internal path: h_cat = [h_pool[c0], h_pool[c1]] ; fused GEMM over combined
// N = 512 (U_f) + 768 (U_iou). Wave owns groups g = 2w, 2w+1; group g needs
// tiles n = g*16 + {0,256}(f0,f1), 512 + g*16 + {0,256,512}(i,o,u).
// Tile order: t = ch*2 + gi, ch in {f0,f1,i,o,u}.
// ---------------------------------------------------------------------------
template <bool USE_WS>
__global__ __launch_bounds__(256) void treelstm_internal(
    const float* __restrict__ h_pool, const float* __restrict__ c_pool,
    const int* __restrict__ cidx,
    const float* __restrict__ Ufw, const float* __restrict__ Ufb,
    const float* __restrict__ Uiw, const float* __restrict__ Uib,
    const __bf16* __restrict__ wsB,
    float* __restrict__ outH, float* __restrict__ outC) {
  __shared__ __attribute__((aligned(16))) __bf16 sA[32][2 * HS + 8];

  const int tid = threadIdx.x;
  const int m0 = blockIdx.x * 32;

  {
    const int r = tid >> 3, seg = tid & 7;
    const int node = m0 + r;
    const int child = seg >> 2;
    const int ci = cidx[node * 2 + child];
    const int sc = (seg & 3) * 64;
    const float4* src = (const float4*)(h_pool + (size_t)ci * HS + sc);
    bf16x2* dst = (bf16x2*)(&sA[r][child * HS + sc]);
#pragma unroll
    for (int q = 0; q < 16; ++q) {
      float4 f = src[q];
      dst[2 * q]     = cvt2(f.x, f.y);
      dst[2 * q + 1] = cvt2(f.z, f.w);
    }
  }
  __syncthreads();

  const int wave = tid >> 5, lane = tid & 31;
  const int nrow = lane & 15, kb = lane >> 4;

  const __bf16* bp16[10] = {};
  const float*  bp32[10] = {};
#pragma unroll
  for (int t = 0; t < 10; ++t) {
    const int gi = t & 1, ch = t >> 1;
    if (USE_WS) {
      const int tile = (wave * 2 + gi) + ch * 16;
      bp16[t] = wsB + ((size_t)tile * 16 * 32 + lane) * 16;
    } else {
      const int n = (wave * 2 + gi) * 16 + ch * 256 + nrow;
      const float* wr = (ch < 2) ? (Ufw + (size_t)n * (2 * HS))
                                 : (Uiw + (size_t)(n - 512) * (2 * HS));
      bp32[t] = wr + kb * 16;
    }
  }
  const __bf16* aBase0 = &sA[nrow][kb * 16];
  const __bf16* aBase1 = &sA[nrow + 16][kb * 16];

  v8f zero;
#pragma unroll
  for (int i = 0; i < 8; ++i) zero[i] = 0.0f;
  v8f acc[10][2];
#pragma unroll
  for (int t = 0; t < 10; ++t) {
    acc[t][0] = zero;
    acc[t][1] = zero;
  }

#pragma unroll
  for (int ks = 0; ks < (2 * HS) / 32; ++ks) {
    v16bf a0 = load_a_lds(aBase0, ks * 32);
    v16bf a1 = load_a_lds(aBase1, ks * 32);
#pragma unroll
    for (int t = 0; t < 10; ++t) {
      v16bf b;
      if (USE_WS) b = *(const v16bf*)(bp16[t] + ks * 512);
      else        b = load_b_f32(bp32[t] + ks * 32);
      acc[t][0] = WMMA_BF16(a0, b, acc[t][0]);
      acc[t][1] = WMMA_BF16(a1, b, acc[t][1]);
    }
  }

#pragma unroll
  for (int gi = 0; gi < 2; ++gi) {
    const int col = (wave * 2 + gi) * 16 + nrow;
    const float bF0 = Ufb[col], bF1 = Ufb[HS + col];
    const float bI = Uib[col], bO = Uib[HS + col], bU = Uib[2 * HS + col];
#pragma unroll
    for (int mt = 0; mt < 2; ++mt) {
#pragma unroll
      for (int v = 0; v < 8; ++v) {
        const int row = m0 + mt * 16 + v + 8 * kb;
        const int e0 = cidx[row * 2], e1 = cidx[row * 2 + 1];
        const float c0 = c_pool[(size_t)e0 * HS + col];
        const float c1 = c_pool[(size_t)e1 * HS + col];
        float cf = sigm(acc[0 * 2 + gi][mt][v] + bF0) * c0 +
                   sigm(acc[1 * 2 + gi][mt][v] + bF1) * c1;
        float cc = sigm(acc[2 * 2 + gi][mt][v] + bI) * tanh_(acc[4 * 2 + gi][mt][v] + bU) + cf;
        float hh = sigm(acc[3 * 2 + gi][mt][v] + bO) * tanh_(cc);
        outH[(size_t)row * HS + col] = hh;
        outC[(size_t)row * HS + col] = cc;
      }
    }
  }
}

extern "C" void kernel_launch(void* const* d_in, const int* in_sizes, int n_in,
                              void* d_out, int out_size, void* d_ws, size_t ws_size,
                              hipStream_t stream) {
  (void)in_sizes; (void)n_in; (void)out_size;
  const float* x      = (const float*)d_in[0];
  const float* h_pool = (const float*)d_in[1];
  const float* c_pool = (const float*)d_in[2];
  const int*   cidx   = (const int*)d_in[3];
  const float* Ww     = (const float*)d_in[4];
  const float* Wb     = (const float*)d_in[5];
  const float* Uiw    = (const float*)d_in[6];
  const float* Uib    = (const float*)d_in[7];
  const float* Ufw    = (const float*)d_in[8];
  const float* Ufb    = (const float*)d_in[9];
  float* out = (float*)d_out;

  dim3 block(256);
  const bool use_ws = (d_ws != nullptr) && (ws_size >= WS_NEEDED);
  if (use_ws) {
    __bf16* wsL = (__bf16*)d_ws;
    __bf16* wsI = (__bf16*)((char*)d_ws + WS_LEAF_BYTES);
    swizzle_leaf<<<12288 / 256, block, 0, stream>>>(Ww, wsL);
    swizzle_internal<<<40960 / 256, block, 0, stream>>>(Ufw, Uiw, wsI);
    treelstm_leaf<true><<<NN / 32, block, 0, stream>>>(x, Ww, Wb, wsL, out, out + NHOFF);
    treelstm_internal<true><<<NN / 32, block, 0, stream>>>(h_pool, c_pool, cidx, Ufw, Ufb,
                                                           Uiw, Uib, wsI,
                                                           out + 2 * NHOFF, out + 3 * NHOFF);
  } else {
    treelstm_leaf<false><<<NN / 32, block, 0, stream>>>(x, Ww, Wb, nullptr, out, out + NHOFF);
    treelstm_internal<false><<<NN / 32, block, 0, stream>>>(h_pool, c_pool, cidx, Ufw, Ufb,
                                                            Uiw, Uib, nullptr,
                                                            out + 2 * NHOFF, out + 3 * NHOFF);
  }
}
